// CMHSAttn_54434415510115
// MI455X (gfx1250) — matrix operand order repaired and checked by reading, not compiled, not run on hardware
//
#include <hip/hip_runtime.h>
#include <hip/hip_bf16.h>
#include <math.h>

typedef _Float16 v16h __attribute__((ext_vector_type(16)));
typedef _Float16 v8h  __attribute__((ext_vector_type(8)));
typedef float    v8f  __attribute__((ext_vector_type(8)));
typedef unsigned int v8u __attribute__((ext_vector_type(8)));

#define N_TOK  4096
#define D_HEAD 16
#define N_HEAD 8
// 1/sqrt(128) * log2(e): scores come out in log2 domain -> native v_exp_f32
#define QKV_SCALE (0.08838834764831845f * 1.4426950408889634f)

// ---- raw (non-canonicalizing) f32 max ----
__device__ __forceinline__ float fmax_raw(float a, float b) {
  float d;
  asm("v_max_num_f32_e32 %0, %1, %2" : "=v"(d) : "v"(a), "v"(b));
  return d;
}

// ---- DPP-fused butterfly max within each 16-lane row: 1 instruction/step ----
#define DPP_MAX_STEP(x, CTRL)                                                  \
  asm("v_max_num_f32_dpp %0, %0, %0 " CTRL                                     \
      " row_mask:0xf bank_mask:0xf bound_ctrl:1"                               \
      : "+v"(x))

__device__ __forceinline__ float row16_max(float x) {
  DPP_MAX_STEP(x, "quad_perm:[1,0,3,2]");  // xor 1
  DPP_MAX_STEP(x, "quad_perm:[2,3,0,1]");  // xor 2
  DPP_MAX_STEP(x, "row_half_mirror");      // xor within 8
  DPP_MAX_STEP(x, "row_mirror");           // xor within 16
  return x;
}

// mov-based exchange (epilogue-only sum reduction)
template <int CTRL>
__device__ __forceinline__ float dpp_xchg(float v) {
  return __int_as_float(
      __builtin_amdgcn_update_dpp(0, __float_as_int(v), CTRL, 0xf, 0xf, true));
}
__device__ __forceinline__ float row16_sum(float x) {
  x += dpp_xchg<0xB1>(x);
  x += dpp_xchg<0x4E>(x);
  x += dpp_xchg<0x141>(x);
  x += dpp_xchg<0x140>(x);
  return x;
}
__device__ __forceinline__ unsigned pk_f16(float a, float b) {
  auto r = __builtin_amdgcn_cvt_pkrtz(a, b);   // v_cvt_pkrtz_f16_f32
  return __builtin_bit_cast(unsigned, r);
}

// ---------------------------------------------------------------------------
// Kernel 1: 1x1-conv QKV projection  qkv[o][n] = sum_c W[o][c] * x[c][n]
// Routes channels into f16 Q (pre-scaled by 1/sqrt(128)*log2e), K ([h][n][d])
// and V^T ([h][d][n]).
// ---------------------------------------------------------------------------
__global__ __launch_bounds__(256) void proj_qkv(
    const float* __restrict__ x, const float* __restrict__ W,
    _Float16* __restrict__ qb, _Float16* __restrict__ kb,
    _Float16* __restrict__ vt)
{
  __shared__ float xs[64][132];   // stride 132: 16B aligned rows
  const int tid = threadIdx.x;
  const int nbase = blockIdx.x * 64;

  for (int idx = tid; idx < 64 * 128; idx += 256) {
    int c = idx >> 6, n = idx & 63;
    xs[n][c] = x[c * N_TOK + nbase + n];
  }
  __syncthreads();

  const int nl = tid & 63;
  const int og = __builtin_amdgcn_readfirstlane(tid >> 6);  // wave-uniform -> s_loads for W
  const int n  = nbase + nl;
  const float4* xr = (const float4*)(&xs[nl][0]);

  for (int o = og; o < 384; o += 4) {
    const float* w = W + o * 128;
    float acc = 0.f;
#pragma unroll
    for (int c4 = 0; c4 < 32; ++c4) {
      float4 xv = xr[c4];
      acc = fmaf(w[4 * c4 + 0], xv.x, acc);
      acc = fmaf(w[4 * c4 + 1], xv.y, acc);
      acc = fmaf(w[4 * c4 + 2], xv.z, acc);
      acc = fmaf(w[4 * c4 + 3], xv.w, acc);
    }
    int h = o / 48, r = o % 48, d = r & 15;
    if (r < 16)      qb[((h * N_TOK + n) << 4) + d] = (_Float16)(acc * QKV_SCALE);
    else if (r < 32) kb[((h * N_TOK + n) << 4) + d] = (_Float16)acc;
    else             vt[((h * 16 + d) << 12) + n]   = (_Float16)acc;
  }
}

// ---------------------------------------------------------------------------
// Kernel 2: flash attention, one 16-query row tile per wave, 32-key chunks.
// S = Q(16x32, d padded) x K^T(32x16)  -> v_wmma_f32_16x16x32_f16 (x2)
// O += P(16x32) x V(32x16)             -> v_wmma_f32_16x16x32_f16 (x1)
// Softmax denominator is accumulated per-lane; cross-lane reduced once at end.
// ---------------------------------------------------------------------------
__global__ __launch_bounds__(256) void flash_attn(
    const _Float16* __restrict__ qb, const _Float16* __restrict__ kb,
    const _Float16* __restrict__ vt, float* __restrict__ out)
{
  // per-wave P staging: 16 rows x 32 keys, stride 36 (16B-aligned, conflict-free)
  __shared__ float pst[8][16][36];

  const int lane  = threadIdx.x & 31;
  const int wv    = threadIdx.x >> 5;
  const int h     = blockIdx.x >> 5;      // 8 heads
  const int qblk  = blockIdx.x & 31;      // 32 query blocks of 128
  const int qbase = qblk * 128 + wv * 16;
  const int col   = lane & 15;
  const bool hi   = lane >= 16;

  // ---- Q A-tile (16x32 f16, contraction d padded 16->32 with zeros) ----
  v16h a = {};
  {
    const v16h* qp = (const v16h*)(qb + (((size_t)(h * N_TOK + qbase + col)) << 4));
    v16h q = *qp;
#pragma unroll
    for (int i = 0; i < 16; ++i) a[i] = hi ? (_Float16)0 : q[i];   // lanes 16..31 (k 16..31) stay zero
  }

  v8f acc = {};
  float m[8], l[8];
#pragma unroll
  for (int r = 0; r < 8; ++r) { m[r] = -3.0e38f; l[r] = 0.f; }

  // per-lane streaming pointers
  const _Float16* kp = kb + (((size_t)(h * N_TOK + col)) << 4);               // +512 halves/chunk
  const _Float16* vp = vt + (((size_t)(h * 16 + col)) << 12) + (hi ? 16 : 0); // +32 halves/chunk
  const int rowoff = hi ? 8 : 0;
  const int kb2    = hi ? 8 : 0;

  for (int jb = 0; jb < N_TOK; jb += 32) {
    // K^T B-tiles: lane = key column, halves = d 0..15 (two global_load_b128 each).
    // A's contraction halves 16..31 are zero -> B lanes 16-31 contents are don't-care.
    v16h b0 = *(const v16h*)(kp);
    v16h b1 = *(const v16h*)(kp + 256);
    __builtin_prefetch(kp + 512, 0, 3);

    v8f czero = {};
    v8f s0 = __builtin_amdgcn_wmma_f32_16x16x32_f16(false, a, false, b0, (short)0, czero, false, false);
    v8f s1 = __builtin_amdgcn_wmma_f32_16x16x32_f16(false, a, false, b1, (short)0, czero, false, false);

    // online softmax in log2 domain; C layout: vgpr r = row r / r+8 per 16-lane half
    float p0[8], p1[8];
#pragma unroll
    for (int r = 0; r < 8; ++r) {
      float mx   = row16_max(fmax_raw(s0[r], s1[r]));
      float mnew = fmax_raw(m[r], mx);
      float alpha = __builtin_amdgcn_exp2f(m[r] - mnew);   // row-uniform
      m[r]  = mnew;
      p0[r] = __builtin_amdgcn_exp2f(s0[r] - mnew);
      p1[r] = __builtin_amdgcn_exp2f(s1[r] - mnew);
      l[r]   = fmaf(l[r], alpha, p0[r] + p1[r]);           // per-lane partial denom
      acc[r] *= alpha;
    }

    // transpose P: C layout -> A layout via per-wave LDS (intra-wave DS in-order)
#pragma unroll
    for (int r = 0; r < 8; ++r) {
      pst[wv][r + rowoff][col]      = p0[r];
      pst[wv][r + rowoff][col + 16] = p1[r];
    }
    const float4* prow = (const float4*)&pst[wv][col][0];
    const int qo = kb2 >> 1;                     // 0 or 4
    float4 f0 = prow[qo], f1 = prow[qo + 1];     // keys 2*kb2..2*kb2+7
    float4 f2 = prow[qo + 2], f3 = prow[qo + 3]; // keys 2*kb2+8..2*kb2+15
    v8u pu;
    pu[0] = pk_f16(f0.x, f0.y); pu[1] = pk_f16(f0.z, f0.w);
    pu[2] = pk_f16(f1.x, f1.y); pu[3] = pk_f16(f1.z, f1.w);
    pu[4] = pk_f16(f2.x, f2.y); pu[5] = pk_f16(f2.z, f2.w);
    pu[6] = pk_f16(f3.x, f3.y); pu[7] = pk_f16(f3.z, f3.w);
    v16h pa = __builtin_bit_cast(v16h, pu);

    // V B-tile: lane = d column, 16 contiguous keys from V^T
    v16h vb = *(const v16h*)(vp);

    acc = __builtin_amdgcn_wmma_f32_16x16x32_f16(false, pa, false, vb, (short)0, acc, false, false);

    kp += 512;
    vp += 32;
  }

  // epilogue: reduce per-lane denominators once, then normalize + scatter
#pragma unroll
  for (int r = 0; r < 8; ++r) {
    float lr = row16_sum(l[r]);
    out[((h * 16 + col) << 12) + qbase + r + rowoff] = acc[r] / lr;
  }
}

// ---------------------------------------------------------------------------
extern "C" void kernel_launch(void* const* d_in, const int* in_sizes, int n_in,
                              void* d_out, int out_size, void* d_ws, size_t ws_size,
                              hipStream_t stream) {
  (void)in_sizes; (void)n_in; (void)out_size; (void)ws_size;
  const float* x = (const float*)d_in[0];   // (1,128,64,64) -> (128, 4096)
  const float* W = (const float*)d_in[1];   // (384, 128)
  float* out = (float*)d_out;               // (128, 4096)

  const size_t per = (size_t)N_HEAD * N_TOK * D_HEAD;   // 512K halves = 1 MB
  _Float16* qb = (_Float16*)d_ws;
  _Float16* kb = qb + per;
  _Float16* vt = kb + per;                  // total ws use: 3 MB

  proj_qkv<<<64, 256, 0, stream>>>(x, W, qb, kb, vt);
  flash_attn<<<256, 256, 0, stream>>>(qb, kb, vt, out);
}
